// MultiHeadAttention_11476152614920
// MI455X (gfx1250) — matrix operand order repaired and checked by reading, not compiled, not run on hardware
//
#include <hip/hip_runtime.h>
#include <hip/hip_bf16.h>

#define D_MODEL 1024
#define SEQ     4096
#define NHEADS  16
#define DK      64

typedef __attribute__((ext_vector_type(16))) __bf16 bf16x16;
typedef __attribute__((ext_vector_type(8)))  __bf16 bf16x8;
typedef __attribute__((ext_vector_type(8)))  float  f32x8;

__device__ __forceinline__ f32x8 wmma_bf16(bf16x16 a, bf16x16 b, f32x8 c) {
  // D = A(16x32 bf16) * B(32x16 bf16) + C(16x16 f32)
  return __builtin_amdgcn_wmma_f32_16x16x32_bf16(
      /*neg_a=*/false, a, /*neg_b=*/false, b,
      /*c_mod=*/(short)0, c, /*reuse_a=*/false, /*reuse_b=*/false);
}

// CDNA5 async global->LDS copy (32-bit), tracked by ASYNCcnt.
// lds_addr is the addrspace(3) byte offset (low 32 bits of the flat pointer).
__device__ __forceinline__ void async_g2l_b32(unsigned int lds_addr, const void* gaddr) {
  asm volatile("global_load_async_to_lds_b32 %0, %1, off"
               :: "v"(lds_addr), "v"(gaddr) : "memory");
}
__device__ __forceinline__ void s_wait_async0() {
  asm volatile("s_wait_asynccnt 0" ::: "memory");
}

// A-fragment 16x32 (MxK), contiguous k order (matches load_b_frag):
// lane (ln, half): row = `row`, element e -> k = k0 + 16*half + e
// => two contiguous 8-element (16B) chunks at k0+half*16 and k0+half*16+8.
__device__ __forceinline__ bf16x16 load_a_frag(const __bf16* __restrict__ base,
                                               int row_stride, int row, int k0, int half) {
  const __bf16* p = base + (size_t)row * row_stride + k0 + half * 16;
  bf16x8 lo = *(const bf16x8*)(p);
  bf16x8 hi = *(const bf16x8*)(p + 8);
  bf16x16 a;
#pragma unroll
  for (int i = 0; i < 8; ++i) { a[i] = lo[i]; a[8 + i] = hi[i]; }
  return a;
}

// B-fragment 32x16 (KxN): lane (ln, half) covers column n, element e -> k = k0 + e + 16*half.
// B[k][n] is sourced from base[n * row_stride + k] (row n of `base` holds the k-line),
// i.e. one contiguous 32B load per lane.
__device__ __forceinline__ bf16x16 load_b_frag(const __bf16* __restrict__ base,
                                               int row_stride, int ncol, int k0, int half) {
  return *(const bf16x16*)(base + (size_t)ncol * row_stride + k0 + 16 * half);
}

// ---------------------------------------------------------------------------
__global__ void cvt_f32_bf16(const float* __restrict__ src, __bf16* __restrict__ dst, int n) {
  int i = blockIdx.x * blockDim.x + threadIdx.x;
  int stride = gridDim.x * blockDim.x;
  for (; i < n; i += stride) dst[i] = (__bf16)src[i];
}

// ---------------------------------------------------------------------------
// C[M,N] = (A[M,K]_bf16 @ W[N,K]_bf16^T + bias[N]) * scale
// mode 0: bf16 row-major out[M,N]
// mode 1: bf16 transposed out[N,M]   (for V -> Vt, packed 16B stores)
// mode 2: f32  row-major out[M,N]
// Block = 128 threads = 4 waves sharing one 16-row M-tile; wave w owns a
// 64-wide N-span. The shared 16x32 A-tile is double-buffered in LDS and
// filled with async global->LDS copies (ASYNCcnt), read back as ds_load_b128.
__global__ void gemm_bias(const __bf16* __restrict__ A, const __bf16* __restrict__ W,
                          const float* __restrict__ bias, void* __restrict__ out,
                          int M, int N, int K, float scale, int mode) {
  __shared__ __align__(16) __bf16 atile[2][16 * 32];

  const int lane = threadIdx.x & 31;
  const int half = lane >> 4;
  const int ln   = lane & 15;
  const int wv   = threadIdx.x >> 5;              // wave in block: 0..3
  const int nblk = N >> 8;                        // number of 256-wide N blocks
  const int m0   = ((int)blockIdx.x / nblk) * 16;
  const int n0   = ((int)blockIdx.x % nblk) * 256 + wv * 64;
  if (m0 >= M) return;  // block-uniform

  // async staging assignment: thread -> (row 0..15, dword chunk 0..7)
  const int tr = threadIdx.x >> 3;
  const int tc = threadIdx.x & 7;
  const __bf16* gA = A + (size_t)(m0 + tr) * K + tc * 2;   // 4B per thread
  const unsigned int lws = (unsigned int)(tr * 64 + tc * 4);  // LDS byte offset in tile

  // prime buffer 0
  async_g2l_b32((unsigned int)(uintptr_t)&atile[0][0] + lws, gA);
  s_wait_async0();
  __syncthreads();

  f32x8 acc[4];
#pragma unroll
  for (int t = 0; t < 4; ++t)
#pragma unroll
    for (int r = 0; r < 8; ++r) acc[t][r] = 0.0f;

  int cur = 0;
  for (int k0 = 0; k0 < K; k0 += 32, cur ^= 1) {
    // kick off async copy of the next A tile into the other buffer
    if (k0 + 32 < K)
      async_g2l_b32((unsigned int)(uintptr_t)&atile[cur ^ 1][0] + lws, gA + k0 + 32);
    // prefetch the B stream 256 k (8 k-blocks) ahead
    if (k0 + 256 < K)
      __builtin_prefetch(W + (size_t)(n0 + ln) * K + k0 + 256, 0, 1);

    // A fragment from LDS (ds_load_b128 x2), all four B fragments, then WMMAs
    bf16x16 a  = load_a_frag(atile[cur], 32, ln, 0, half);
    bf16x16 b0 = load_b_frag(W, K, n0 +  0 + ln, k0, half);
    bf16x16 b1 = load_b_frag(W, K, n0 + 16 + ln, k0, half);
    bf16x16 b2 = load_b_frag(W, K, n0 + 32 + ln, k0, half);
    bf16x16 b3 = load_b_frag(W, K, n0 + 48 + ln, k0, half);
    acc[0] = wmma_bf16(a, b0, acc[0]);
    acc[1] = wmma_bf16(a, b1, acc[1]);
    acc[2] = wmma_bf16(a, b2, acc[2]);
    acc[3] = wmma_bf16(a, b3, acc[3]);

    // next buffer must be complete (and visible block-wide) before next iter
    s_wait_async0();
    __syncthreads();
  }

#pragma unroll
  for (int t = 0; t < 4; ++t) {
    const int n  = n0 + t * 16 + ln;       // this lane's column
    const float bn = bias[n];
    if (mode == 1) {                        // transposed bf16: out[n][m0+half*8 .. +7]
      bf16x8 v;
#pragma unroll
      for (int r = 0; r < 8; ++r) v[r] = (__bf16)((acc[t][r] + bn) * scale);
      *(bf16x8*)((__bf16*)out + (size_t)n * M + m0 + half * 8) = v;
    } else if (mode == 0) {                 // row-major bf16
#pragma unroll
      for (int r = 0; r < 8; ++r)
        ((__bf16*)out)[(size_t)(m0 + r + 8 * half) * N + n] =
            (__bf16)((acc[t][r] + bn) * scale);
    } else {                                // row-major f32
#pragma unroll
      for (int r = 0; r < 8; ++r)
        ((float*)out)[(size_t)(m0 + r + 8 * half) * N + n] = (acc[t][r] + bn) * scale;
    }
  }
}

// ---------------------------------------------------------------------------
// Flash attention, transposed-score formulation.
// One wave per (head, 16-query tile). Q is pre-scaled by log2(e)/sqrt(dk).
// S^T[kk][q] = K[kk,:]·Q[q,:]   (M=key block, N=query, Kdim=d)
// O^T[d][q]  = sum_kk Vt[d][kk] · P^T[kk][q]
__global__ void attn(const __bf16* __restrict__ Qb, const __bf16* __restrict__ Kb,
                     const __bf16* __restrict__ Vt, __bf16* __restrict__ ctx) {
  const int lane = threadIdx.x & 31;
  const int half = lane >> 4;
  const int ln   = lane & 15;
  const int wave = blockIdx.x * (blockDim.x >> 5) + (threadIdx.x >> 5);
  const int head = wave >> 8;         // 256 query tiles per head
  const int qt   = wave & 255;
  if (head >= NHEADS) return;         // wave-uniform
  const int q0    = qt * 16;
  const int hbase = head * DK;

  // Q B-fragments: column = query (ln), kdim = d; contiguous reads of Q rows.
  const bf16x16 bq0 = load_b_frag(Qb, D_MODEL, q0 + ln, hbase +  0, half);
  const bf16x16 bq1 = load_b_frag(Qb, D_MODEL, q0 + ln, hbase + 32, half);

  f32x8 acc[4];
#pragma unroll
  for (int t = 0; t < 4; ++t)
#pragma unroll
    for (int r = 0; r < 8; ++r) acc[t][r] = 0.0f;

  float mprev = -1e30f;
  float lsum  = 0.0f;

  for (int kk = 0; kk < SEQ; kk += 32) {
    // --- scores for 32 keys: two 16x16 C-tiles (rows = keys, cols = queries)
    f32x8 s0 = {0,0,0,0,0,0,0,0};
    f32x8 s1 = {0,0,0,0,0,0,0,0};
    {
      bf16x16 ka0 = load_a_frag(Kb, D_MODEL, kk + ln,      hbase +  0, half);
      bf16x16 ka1 = load_a_frag(Kb, D_MODEL, kk + ln,      hbase + 32, half);
      bf16x16 ka2 = load_a_frag(Kb, D_MODEL, kk + 16 + ln, hbase +  0, half);
      bf16x16 ka3 = load_a_frag(Kb, D_MODEL, kk + 16 + ln, hbase + 32, half);
      s0 = wmma_bf16(ka0, bq0, s0);
      s0 = wmma_bf16(ka1, bq1, s0);
      s1 = wmma_bf16(ka2, bq0, s1);
      s1 = wmma_bf16(ka3, bq1, s1);
    }

    // --- online softmax over this lane's column (q = ln). Scores are already
    // scaled by log2(e)/sqrt(dk), so exp2 gives exact softmax weights.
    float mloc = -1e30f;
#pragma unroll
    for (int r = 0; r < 8; ++r) mloc = fmaxf(mloc, fmaxf(s0[r], s1[r]));
    mloc = fmaxf(mloc, __shfl_xor(mloc, 16, 32));   // combine the two lane halves
    const float mnew = fmaxf(mprev, mloc);
    const float corr = exp2f(mprev - mnew);

    float p0[8], p1[8];
    float sloc = 0.0f;
#pragma unroll
    for (int r = 0; r < 8; ++r) {
      p0[r] = exp2f(s0[r] - mnew);
      p1[r] = exp2f(s1[r] - mnew);
      sloc += p0[r] + p1[r];
    }
    sloc += __shfl_xor(sloc, 16, 32);
    lsum = lsum * corr + sloc;
#pragma unroll
    for (int t = 0; t < 4; ++t)
#pragma unroll
      for (int r = 0; r < 8; ++r) acc[t][r] *= corr;
    mprev = mnew;

    // --- convert the two P^T C-tiles to a 32x16 B-fragment in-register.
    // B element e at lane (ln,half) must be P^T[e + 16*half][ln]:
    //   half0: e<8 -> own p0[e];       e>=8 -> other half's p0[e-8]
    //   half1: e<8 -> other's p1[e];   e>=8 -> own p1[e-8]
    bf16x16 pB;
#pragma unroll
    for (int r = 0; r < 8; ++r) {
      const float o0 = __shfl_xor(p0[r], 16, 32);
      const float o1 = __shfl_xor(p1[r], 16, 32);
      pB[r]     = (__bf16)(half ? o1    : p0[r]);
      pB[8 + r] = (__bf16)(half ? p1[r] : o0);
    }

    // --- O^T += Vt-block * P^T  (A rows = d, contiguous reads of Vt rows)
#pragma unroll
    for (int t = 0; t < 4; ++t) {
      bf16x16 av = load_a_frag(Vt, SEQ, hbase + t * 16 + ln, kk, half);
      acc[t] = wmma_bf16(av, pB, acc[t]);
    }
  }

  // --- normalize and store ctx (row-major bf16 [SEQ, D_MODEL], packed 16B)
  const float inv = 1.0f / lsum;
#pragma unroll
  for (int t = 0; t < 4; ++t) {
    bf16x8 v;
#pragma unroll
    for (int r = 0; r < 8; ++r) v[r] = (__bf16)(acc[t][r] * inv);
    *(bf16x8*)(ctx + (size_t)(q0 + ln) * D_MODEL + hbase + t * 16 + 8 * half) = v;
  }
}

// ---------------------------------------------------------------------------
extern "C" void kernel_launch(void* const* d_in, const int* in_sizes, int n_in,
                              void* d_out, int out_size, void* d_ws, size_t ws_size,
                              hipStream_t stream) {
  (void)in_sizes; (void)n_in; (void)out_size; (void)ws_size;
  const float* x   = (const float*)d_in[0];
  const float* W_q = (const float*)d_in[1];
  const float* b_q = (const float*)d_in[2];
  const float* W_k = (const float*)d_in[3];
  const float* b_k = (const float*)d_in[4];
  const float* W_v = (const float*)d_in[5];
  const float* b_v = (const float*)d_in[6];
  const float* W_o = (const float*)d_in[7];
  const float* b_o = (const float*)d_in[8];
  float* out = (float*)d_out;

  char* ws = (char*)d_ws;
  const size_t MB = (size_t)1 << 20;
  __bf16* xb   = (__bf16*)(ws +  0 * MB);  // 8 MB
  __bf16* Wqb  = (__bf16*)(ws +  8 * MB);  // 2 MB
  __bf16* Wkb  = (__bf16*)(ws + 10 * MB);
  __bf16* Wvb  = (__bf16*)(ws + 12 * MB);
  __bf16* Wob  = (__bf16*)(ws + 14 * MB);
  __bf16* Qb   = (__bf16*)(ws + 16 * MB);  // 8 MB, pre-scaled by log2e/8
  __bf16* Kb   = (__bf16*)(ws + 24 * MB);  // 8 MB
  __bf16* Vtb  = (__bf16*)(ws + 32 * MB);  // 8 MB, transposed [D_MODEL][SEQ]
  __bf16* ctxb = (__bf16*)(ws + 40 * MB);  // 8 MB

  // 1) f32 -> bf16 conversions
  cvt_f32_bf16<<<2048, 256, 0, stream>>>(x,   xb,  SEQ * D_MODEL);
  cvt_f32_bf16<<<1024, 256, 0, stream>>>(W_q, Wqb, D_MODEL * D_MODEL);
  cvt_f32_bf16<<<1024, 256, 0, stream>>>(W_k, Wkb, D_MODEL * D_MODEL);
  cvt_f32_bf16<<<1024, 256, 0, stream>>>(W_v, Wvb, D_MODEL * D_MODEL);
  cvt_f32_bf16<<<1024, 256, 0, stream>>>(W_o, Wob, D_MODEL * D_MODEL);

  // 2) projections (fold softmax scale * log2e into Q, incl. its bias)
  const float qscale = 1.4426950408889634f / 8.0f;  // log2(e)/sqrt(64)
  const int gemm_blocks = (SEQ / 16) * (D_MODEL / 256);  // 4 waves/block, shared M-tile
  gemm_bias<<<gemm_blocks, 128, 0, stream>>>(xb, Wqb, b_q, Qb,  SEQ, D_MODEL, D_MODEL, qscale, 0);
  gemm_bias<<<gemm_blocks, 128, 0, stream>>>(xb, Wkb, b_k, Kb,  SEQ, D_MODEL, D_MODEL, 1.0f,   0);
  gemm_bias<<<gemm_blocks, 128, 0, stream>>>(xb, Wvb, b_v, Vtb, SEQ, D_MODEL, D_MODEL, 1.0f,   1);

  // 3) flash attention: 16 heads x 256 query tiles = 4096 waves
  attn<<<(NHEADS * (SEQ / 16)) / 4, 128, 0, stream>>>(Qb, Kb, Vtb, ctxb);

  // 4) output projection, f32 result
  gemm_bias<<<gemm_blocks, 128, 0, stream>>>(ctxb, Wob, b_o, out, SEQ, D_MODEL, D_MODEL, 1.0f, 2);
}
